// SpatialAttentionHead_62036507623915
// MI455X (gfx1250) — matrix operand order repaired and checked by reading, not compiled, not run on hardware
//
#include <hip/hip_runtime.h>
#include <hip/hip_bf16.h>
#include <cstdint>

// SpatialAttentionHead for MI455X (gfx1250), bf16 WMMA pipeline with fp32 accumulate.
// B=16, H=256, C=128, W=128.  Requires ws_size >= 604 MB (4*128MB + 64MB).

typedef __attribute__((ext_vector_type(16))) __bf16 v16bf;
typedef __attribute__((ext_vector_type(8)))  __bf16 v8bf;
typedef __attribute__((ext_vector_type(8)))  float  v8f;
typedef unsigned short u16;

constexpr int NB = 16, NH = 256, NC = 128, NW = 128;
constexpr int RS   = NC + 8;   // 136 halves: padded LDS row stride (K = 128), 16B-aligned rows
constexpr int RS64 = 64 + 8;   // 72 halves:  padded row for 64-wide K chunks

__device__ __forceinline__ u16 f2bf(float f) {
  union { float f; unsigned u; } v; v.f = f;
  unsigned u = v.u;
  return (u16)((u + 0x7FFFu + ((u >> 16) & 1u)) >> 16);  // round-to-nearest-even
}

// Async global->LDS 16B copy (CDNA5 GLOBAL_LOAD_ASYNC_TO_LDS_B128, ASYNCcnt-tracked).
// GV mode: 64-bit per-lane global address, per-lane LDS byte offset.
__device__ __forceinline__ void async_copy16(unsigned ldsOff, const void* gaddr) {
  asm volatile("global_load_async_to_lds_b128 %0, %1, off"
               :: "v"(ldsOff), "v"(gaddr) : "memory");
}
__device__ __forceinline__ void async_wait0() {
  asm volatile("s_wait_asynccnt 0x0" ::: "memory");
}
__device__ __forceinline__ unsigned lds_off(const void* p) {
  return (unsigned)(uintptr_t)p;  // LDS offset lives in addr[31:0] (ISA 10.2 aperture rules)
}

// A fragment 16x32 bf16 (ISA 7.12.2): lane m = l&15; lanes<16 hold K {0..7,16..23},
// lanes>=16 hold K {8..15,24..31}.  Two 16B ds loads per lane.
__device__ __forceinline__ v16bf ldsA(const u16* p, int row, int rs, int k0, int lane) {
  int m  = row + (lane & 15);
  int kb = (lane >> 4) << 3;  // 0 or 8
  v8bf lo = *(const v8bf*)(p + m * rs + k0 + kb);
  v8bf hi = *(const v8bf*)(p + m * rs + k0 + 16 + kb);
  v16bf a;
#pragma unroll
  for (int i = 0; i < 8; ++i) { a[i] = lo[i]; a[i + 8] = hi[i]; }
  return a;
}

// B fragment 32x16 bf16: lane n = l&15; lanes<16 hold K {0..7,16..23}, lanes>=16 K {8..15,24..31}.
// LDS operand stored [n][k]; two 16B ds loads per lane.
__device__ __forceinline__ v16bf ldsB(const u16* p, int row, int rs, int k0, int lane) {
  int n  = row + (lane & 15);
  int kb = (lane >> 4) << 3;  // 0 or 8
  v8bf lo = *(const v8bf*)(p + n * rs + k0 + kb);
  v8bf hi = *(const v8bf*)(p + n * rs + k0 + 16 + kb);
  v16bf b;
#pragma unroll
  for (int i = 0; i < 8; ++i) { b[i] = lo[i]; b[i + 8] = hi[i]; }
  return b;
}

// ---------------- Kernel 1: fused q/k/v 1x1 convs, per (b,h) block ----------------
// out layout (B,W,H,C) bf16, written via LDS transpose for 256B coalesced runs.
__global__ void qkv_kernel(const float* __restrict__ x,
                           const float* __restrict__ Wq, const float* __restrict__ bq,
                           const float* __restrict__ Wk, const float* __restrict__ bk,
                           const float* __restrict__ Wv, const float* __restrict__ bv,
                           u16* __restrict__ qo, u16* __restrict__ ko, u16* __restrict__ vo) {
  extern __shared__ u16 sm[];
  u16* xs = sm;                 // B operand [w][c], RS
  u16* wt = sm + NW * RS;       // A operand [d][c] = W^T, RS
  u16* tb = wt + NC * RS;       // epilogue transpose buffer [w][d], RS
  const int tid = threadIdx.x, lane = tid & 31, wid = tid >> 5;
  const int b = blockIdx.x / NH, h = blockIdx.x % NH;

  const float* xp = x + (size_t)blockIdx.x * NC * NW;   // x[b][h][c][w], w contiguous
  for (int i = tid; i < NC * NW; i += 256) {
    int c = i >> 7, w = i & 127;
    xs[w * RS + c] = f2bf(xp[i]);
  }
  const float* Ws[3] = {Wq, Wk, Wv};
  const float* bs[3] = {bq, bk, bv};
  u16*         os[3] = {qo, ko, vo};
  const v8f vzero = {};
#pragma unroll 1
  for (int mi = 0; mi < 3; ++mi) {
    __syncthreads();
    const float* Wp = Ws[mi];
    for (int i = tid; i < NC * NC; i += 256) {   // W[c][d] -> wt[d][c]
      int c = i >> 7, d = i & 127;
      wt[d * RS + c] = f2bf(Wp[i]);
    }
    __syncthreads();
    v8f acc[8];
#pragma unroll
    for (int nt = 0; nt < 8; ++nt) acc[nt] = vzero;
#pragma unroll
    for (int kk = 0; kk < 4; ++kk) {
      v16bf a = ldsA(wt, wid * 16, RS, kk * 32, lane);
#pragma unroll
      for (int nt = 0; nt < 8; ++nt) {
        v16bf bm = ldsB(xs, nt * 16, RS, kk * 32, lane);
        acc[nt] = __builtin_amdgcn_wmma_f32_16x16x32_bf16(false, a, false, bm,
                                                          (short)0, acc[nt], false, false);
      }
    }
    const float* bp = bs[mi];
    const int dro = wid * 16 + ((lane >> 4) << 3);   // D-tile row: m = r + (lane<16?0:8)
#pragma unroll
    for (int nt = 0; nt < 8; ++nt) {
      int w = nt * 16 + (lane & 15);
#pragma unroll
      for (int r = 0; r < 8; ++r) {
        int d = dro + r;
        tb[w * RS + d] = f2bf(acc[nt][r] + bp[d]);
      }
    }
    __syncthreads();
    u16* op = os[mi] + ((size_t)b * NW) * NH * NC + (size_t)h * NC;  // [b][w][h][c]
    for (int i = tid; i < NW * NC; i += 256) {
      int w = i >> 7, d = i & 127;
      op[(size_t)w * NH * NC + d] = tb[w * RS + d];
    }
  }
}

// ---------------- Kernel 2: scores[b][w][c][d] = (q.kT over H)/sqrt(W), per (b,w) ----------------
__global__ void scores_kernel(const u16* __restrict__ q, const u16* __restrict__ k,
                              float* __restrict__ sc) {
  extern __shared__ u16 sm[];
  u16* qs = sm;              // [c][hh] RS64 (A operand, transposed on copy-in)
  u16* ks = sm + NC * RS64;  // [d][hh] RS64 (B operand, transposed on copy-in)
  const int tid = threadIdx.x, lane = tid & 31, wid = tid >> 5;
  const u16* qp = q + (size_t)blockIdx.x * NH * NC;  // slice [h][c], c contiguous
  const u16* kp = k + (size_t)blockIdx.x * NH * NC;
  const v8f vzero = {};
  v8f acc[8];
#pragma unroll
  for (int nt = 0; nt < 8; ++nt) acc[nt] = vzero;

  for (int hc = 0; hc < 4; ++hc) {        // K = H = 256 in 4 chunks of 64
    __syncthreads();
    const int h0 = hc * 64;
    for (int i = tid; i < 64 * 64; i += 256) {   // coalesced u32 reads, transposed LDS writes
      int hh = i >> 6, cp = i & 63;
      unsigned uq = *(const unsigned*)(qp + (size_t)(h0 + hh) * NC + 2 * cp);
      qs[(2 * cp)     * RS64 + hh] = (u16)uq;
      qs[(2 * cp + 1) * RS64 + hh] = (u16)(uq >> 16);
      unsigned uk = *(const unsigned*)(kp + (size_t)(h0 + hh) * NC + 2 * cp);
      ks[(2 * cp)     * RS64 + hh] = (u16)uk;
      ks[(2 * cp + 1) * RS64 + hh] = (u16)(uk >> 16);
    }
    __syncthreads();
#pragma unroll
    for (int kk = 0; kk < 2; ++kk) {
      v16bf a = ldsA(qs, wid * 16, RS64, kk * 32, lane);
#pragma unroll
      for (int nt = 0; nt < 8; ++nt) {
        v16bf bm = ldsB(ks, nt * 16, RS64, kk * 32, lane);
        acc[nt] = __builtin_amdgcn_wmma_f32_16x16x32_bf16(false, a, false, bm,
                                                          (short)0, acc[nt], false, false);
      }
    }
  }
  float* sp = sc + (size_t)blockIdx.x * NC * NC;   // [c][d], d contiguous -> coalesced
  const float scale = 0.08838834764831845f;        // 1/sqrt(128)
  const int cro = wid * 16 + ((lane >> 4) << 3);
#pragma unroll
  for (int nt = 0; nt < 8; ++nt) {
    int d = nt * 16 + (lane & 15);
#pragma unroll
    for (int r = 0; r < 8; ++r)
      sp[(size_t)(cro + r) * NC + d] = acc[nt][r] * scale;
  }
}

// ---------------- Kernel 3: softmax over W (axis=-1), thread = d, block = (c,b) ----------------
__global__ void softmax_kernel(const float* __restrict__ sc, u16* __restrict__ at) {
  const int d = threadIdx.x;                 // 128 threads
  const int c = blockIdx.x, b = blockIdx.y;
  const size_t base   = ((size_t)b * NW * NC + c) * NC + d;
  const size_t stride = (size_t)NC * NC;     // step in w
  float s[NW];
#pragma unroll
  for (int w = 0; w < NW; ++w) s[w] = sc[base + (size_t)w * stride];
  float mx = -3.4e38f;
#pragma unroll
  for (int w = 0; w < NW; ++w) mx = fmaxf(mx, s[w]);
  float sum = 0.f;
#pragma unroll
  for (int w = 0; w < NW; ++w) { s[w] = __expf(s[w] - mx); sum += s[w]; }
  float inv = 1.f / sum;
#pragma unroll
  for (int w = 0; w < NW; ++w) at[base + (size_t)w * stride] = f2bf(s[w] * inv);
}

// ---------------- Kernel 4: wn[b][w][c][e] = attn x Wn + bn, per (b,w) ----------------
__global__ void wn_kernel(const u16* __restrict__ at, const float* __restrict__ Wn,
                          const float* __restrict__ bn, u16* __restrict__ wno) {
  extern __shared__ u16 sm[];
  u16* as = sm;             // A operand [c][d], RS (identity copy -> async DMA)
  u16* wt = sm + NC * RS;   // B operand [e][d] = Wn^T, RS
  float* bnl = (float*)(wt + NC * RS);
  const int tid = threadIdx.x, lane = tid & 31, wid = tid >> 5;
  const u16* ap = at + (size_t)blockIdx.x * NC * NC;
  // 128 rows x 256B, repacked to 272B LDS rows via per-lane async B128 copies.
  for (int j = tid; j < NC * 16; j += 256) {
    int c = j >> 4, col = j & 15;
    async_copy16(lds_off(as + c * RS) + col * 16, ap + c * NC + col * 8);
  }
  for (int i = tid; i < NC * NC; i += 256) {  // Wn[d][e] -> wt[e][d]
    int dd = i >> 7, e = i & 127;
    wt[e * RS + dd] = f2bf(Wn[i]);
  }
  if (tid < NC) bnl[tid] = bn[tid];
  async_wait0();
  __syncthreads();
  const v8f vzero = {};
  v8f acc[8];
#pragma unroll
  for (int nt = 0; nt < 8; ++nt) acc[nt] = vzero;
#pragma unroll
  for (int kk = 0; kk < 4; ++kk) {
    v16bf a = ldsA(as, wid * 16, RS, kk * 32, lane);
#pragma unroll
    for (int nt = 0; nt < 8; ++nt) {
      v16bf bm = ldsB(wt, nt * 16, RS, kk * 32, lane);
      acc[nt] = __builtin_amdgcn_wmma_f32_16x16x32_bf16(false, a, false, bm,
                                                        (short)0, acc[nt], false, false);
    }
  }
  u16* op = wno + (size_t)blockIdx.x * NC * NC;
  const int cro = wid * 16 + ((lane >> 4) << 3);
#pragma unroll
  for (int nt = 0; nt < 8; ++nt) {
    int e = nt * 16 + (lane & 15);
#pragma unroll
    for (int r = 0; r < 8; ++r)
      op[(size_t)(cro + r) * NC + e] = f2bf(acc[nt][r] + bnl[e]);
  }
}

// ---------------- Kernel 5: m[b][w][c][h] = wn x v (contract d), per (b,w) ----------------
__global__ void av_kernel(const u16* __restrict__ wns, const u16* __restrict__ vqk,
                          u16* __restrict__ mo) {
  extern __shared__ u16 sm[];
  u16* as = sm;             // [c][d] RS (identity copy -> async DMA)
  u16* vs = sm + NC * RS;   // [h][d] RS (identity copy -> async DMA)
  const int tid = threadIdx.x, lane = tid & 31, wid = tid >> 5;
  const u16* ap = wns + (size_t)blockIdx.x * NC * NC;
  const u16* vp = vqk + (size_t)blockIdx.x * NH * NC;
  for (int j = tid; j < NC * 16; j += 256) {
    int c = j >> 4, col = j & 15;
    async_copy16(lds_off(as + c * RS) + col * 16, ap + c * NC + col * 8);
  }
  for (int j = tid; j < NH * 16; j += 256) {
    int hh = j >> 4, col = j & 15;
    async_copy16(lds_off(vs + hh * RS) + col * 16, vp + (size_t)hh * NC + col * 8);
  }
  async_wait0();
  __syncthreads();
  const v8f vzero = {};
  v8f acc[16];
#pragma unroll
  for (int nt = 0; nt < 16; ++nt) acc[nt] = vzero;
#pragma unroll
  for (int kk = 0; kk < 4; ++kk) {
    v16bf a = ldsA(as, wid * 16, RS, kk * 32, lane);
#pragma unroll
    for (int nt = 0; nt < 16; ++nt) {
      v16bf bm = ldsB(vs, nt * 16, RS, kk * 32, lane);
      acc[nt] = __builtin_amdgcn_wmma_f32_16x16x32_bf16(false, a, false, bm,
                                                        (short)0, acc[nt], false, false);
    }
  }
  u16* op = mo + (size_t)blockIdx.x * NC * NH;   // [c][h], h contiguous
  const int cro = wid * 16 + ((lane >> 4) << 3);
#pragma unroll
  for (int nt = 0; nt < 16; ++nt) {
    int hh = nt * 16 + (lane & 15);
#pragma unroll
    for (int r = 0; r < 8; ++r)
      op[(size_t)(cro + r) * NH + hh] = f2bf(acc[nt][r]);
  }
}

// ---------------- Kernel 6: out[b][h][o][w] = Wo^T x m + bo, w-grouped for coalesced stores ----
__global__ void out_kernel(const u16* __restrict__ m, const float* __restrict__ Wo,
                           const float* __restrict__ bo, float* __restrict__ out) {
  extern __shared__ u16 sm[];
  u16* wot = sm;                         // [o][c] = Wo^T, RS
  u16* msl = sm + NC * RS;               // [h16][c], RS
  float* obuf = (float*)(msl + 16 * RS); // [o][h][w] 128x16x16 f32 (128KB, big WGP LDS)
  float* bol  = obuf + NC * 16 * 16;
  const int tid = threadIdx.x, lane = tid & 31, wid = tid >> 5;
  const int w0 = blockIdx.x * 16, h0 = blockIdx.y * 16, b = blockIdx.z;
  for (int i = tid; i < NC * NC; i += 256) {   // Wo[c][o] -> wot[o][c]
    int c = i >> 7, o = i & 127;
    wot[o * RS + c] = f2bf(Wo[i]);
  }
  if (tid < NC) bol[tid] = bo[tid];
#pragma unroll 1
  for (int wl = 0; wl < 16; ++wl) {
    __syncthreads();
    const u16* mp = m + ((size_t)b * NW + (w0 + wl)) * NC * NH;  // [c][h]
    for (int i = tid; i < NC * 8; i += 256) {   // 16-h chunk, transposed into LDS [h][c]
      int c = i >> 3, hp = i & 7;
      unsigned u = *(const unsigned*)(mp + (size_t)c * NH + h0 + 2 * hp);
      msl[(2 * hp)     * RS + c] = (u16)u;
      msl[(2 * hp + 1) * RS + c] = (u16)(u >> 16);
    }
    __syncthreads();
    v8f acc = {};
#pragma unroll
    for (int kk = 0; kk < 4; ++kk) {
      v16bf a  = ldsA(wot, wid * 16, RS, kk * 32, lane);
      v16bf bm = ldsB(msl, 0, RS, kk * 32, lane);
      acc = __builtin_amdgcn_wmma_f32_16x16x32_bf16(false, a, false, bm,
                                                    (short)0, acc, false, false);
    }
    const int oro = wid * 16 + ((lane >> 4) << 3);
    const int hh = lane & 15;
#pragma unroll
    for (int r = 0; r < 8; ++r) {
      int o = oro + r;
      obuf[(o * 16 + hh) * 16 + wl] = acc[r] + bol[o];
    }
  }
  __syncthreads();
  for (int i = tid; i < NC * 16 * 16; i += 256) {  // 64B w-runs, fully buffered
    int wl = i & 15, hh = (i >> 4) & 15, o = i >> 8;
    out[(((size_t)b * NH + (h0 + hh)) * NC + o) * NW + (w0 + wl)] = obuf[i];
  }
}

extern "C" void kernel_launch(void* const* d_in, const int* in_sizes, int n_in,
                              void* d_out, int out_size, void* d_ws, size_t ws_size,
                              hipStream_t stream) {
  (void)in_sizes; (void)n_in; (void)out_size; (void)ws_size;
  const float* x  = (const float*)d_in[0];
  const float* Wq = (const float*)d_in[1];
  const float* bq = (const float*)d_in[2];
  const float* Wk = (const float*)d_in[3];
  const float* bk = (const float*)d_in[4];
  const float* Wv = (const float*)d_in[5];
  const float* bv = (const float*)d_in[6];
  const float* Wn = (const float*)d_in[7];
  const float* bn = (const float*)d_in[8];
  const float* Wo = (const float*)d_in[9];
  const float* bo = (const float*)d_in[10];
  float* out = (float*)d_out;

  char* ws = (char*)d_ws;
  const size_t QSZ = (size_t)NB * NW * NH * NC * 2;  // 128 MB per bf16 tensor
  u16*   q  = (u16*)(ws);
  u16*   k  = (u16*)(ws + QSZ);
  u16*   v  = (u16*)(ws + 2 * QSZ);
  float* sc = (float*)(ws + 3 * QSZ);                // scores f32 (B,W,C,C) = 128 MB
  u16*   at = (u16*)(ws + 4 * QSZ);                  // attn bf16 = 64 MB
  u16*   wn = (u16*)(ws + 3 * QSZ);                  // reuse scores region after softmax
  u16*   mm = (u16*)(ws);                            // reuse q region after scores

  qkv_kernel    <<<NB * NH, 256, (NW * RS + NC * RS + NW * RS) * 2, stream>>>(
      x, Wq, bq, Wk, bk, Wv, bv, q, k, v);
  scores_kernel <<<NB * NW, 256, 2 * NC * RS64 * 2, stream>>>(q, k, sc);
  softmax_kernel<<<dim3(NC, NB), 128, 0, stream>>>(sc, at);
  wn_kernel     <<<NB * NW, 256, 2 * NC * RS * 2 + NC * 4, stream>>>(at, Wn, bn, wn);
  av_kernel     <<<NB * NW, 256, (NC * RS + NH * RS) * 2, stream>>>(wn, v, mm);
  out_kernel    <<<dim3(NW / 16, NH / 16, NB), 256,
                   (NC * RS + 16 * RS) * 2 + (NC * 16 * 16 + NC) * 4, stream>>>(
      mm, Wo, bo, out);
}
